// AttentionHead_67654324847273
// MI455X (gfx1250) — matrix operand order repaired and checked by reading, not compiled, not run on hardware
//
#include <hip/hip_runtime.h>
#include <hip/hip_bf16.h>

#define NB 8
#define LQ 2048
#define SK 2048
#define EE 1024
#define DD 64

typedef __attribute__((ext_vector_type(16))) __bf16 v16bf;
typedef __attribute__((ext_vector_type(8)))  __bf16 v8bf;
typedef __attribute__((ext_vector_type(8)))  float  v8f;

#if __has_builtin(__builtin_amdgcn_sched_group_barrier)
#define SCHED_GROUP(mask, size, id) __builtin_amdgcn_sched_group_barrier(mask, size, id)
#else
#define SCHED_GROUP(mask, size, id)
#endif

// butterfly shuffle within the wave (wave32); masks <16 keep it inside each 16-lane half
static __device__ __forceinline__ float xor_f(float x, int mask) {
    const int lane = threadIdx.x & 31;
    const int idx  = ((lane ^ mask) << 2);
    int r = __builtin_amdgcn_ds_bpermute(idx, __builtin_bit_cast(int, x));
    return __builtin_bit_cast(float, r);
}

// ---------------------------------------------------------------------------
// Kernel 0: transpose + convert the three [E,D] fp32 weights into bf16
// WT[p][D][E] so projection B-fragments become contiguous 16B loads.
// ---------------------------------------------------------------------------
__global__ __launch_bounds__(256) void wtrans_kernel(
    const float* __restrict__ Wq, const float* __restrict__ Wk,
    const float* __restrict__ Wv, __bf16* __restrict__ WT)
{
    const int idx = blockIdx.x * blockDim.x + threadIdx.x;   // 3*1024*64 threads
    if (idx >= 3 * EE * DD) return;
    const int p   = idx / (EE * DD);
    const int rem = idx % (EE * DD);
    const int e   = rem / DD;       // E index (row of W)
    const int d   = rem % DD;       // D index (col of W) -> coalesced read
    const float* W = (p == 0 ? Wq : (p == 1 ? Wk : Wv));
    WT[(size_t)p * DD * EE + (size_t)d * EE + e] = (__bf16)W[rem];
}

// ---------------------------------------------------------------------------
// Kernel 1: fused Q/K/V projection, fp32 -> bf16, WMMA 16x16x32 bf16.
// One wave computes one 16x64 output stripe; per K-step the schedule is pinned
// to: 12 VMEM reads -> converts -> 4 WMMAs, so all loads are in flight at once.
// vh is stored transposed [D,S] with packed 16B stores.
// ---------------------------------------------------------------------------
__global__ __launch_bounds__(256) void proj_kernel(
    const float* __restrict__ q, const float* __restrict__ k, const float* __restrict__ v,
    const __bf16* __restrict__ WT,
    __bf16* __restrict__ qh, __bf16* __restrict__ kh, __bf16* __restrict__ vT)
{
    const int wave = (blockIdx.x * blockDim.x + threadIdx.x) >> 5;
    const int lane = threadIdx.x & 31;
    const int nl   = lane & 15;
    const int half = lane >> 4;

    int t = wave;
    const int rt = t & 127; t >>= 7;   // 128 row tiles (L=2048)
    const int n  = t & 7;   t >>= 3;   // 8 batches
    const int p  = t;                   // 0=Q,1=K,2=V
    if (p >= 3) return;

    const float*  X  = (p == 0 ? q : (p == 1 ? k : v)) + (size_t)n * LQ * EE;
    const __bf16* Wp = WT + (size_t)p * DD * EE;       // [64][1024] bf16

    const int r0 = rt * 16;
    const size_t arow = (size_t)(r0 + nl) * EE;

    v8f acc[4] = {};
    for (int ko = 0; ko < EE; ko += 32) {
        // A fragment: 16-bit A 16x32 layout. Lane half h, row m=nl holds
        // K = h*8+e (e<8) and K = 16+h*8+(e-8) (e>=8): two contiguous 8-float runs.
        v16bf a;
        {
            const float4* p0 = (const float4*)(X + arow + ko + half * 8);
            const float4* p1 = (const float4*)(X + arow + ko + 16 + half * 8);
            float4 x0 = p0[0], x1 = p0[1], x2 = p1[0], x3 = p1[1];
            a[0]=(__bf16)x0.x; a[1]=(__bf16)x0.y; a[2]=(__bf16)x0.z; a[3]=(__bf16)x0.w;
            a[4]=(__bf16)x1.x; a[5]=(__bf16)x1.y; a[6]=(__bf16)x1.z; a[7]=(__bf16)x1.w;
            a[8]=(__bf16)x2.x; a[9]=(__bf16)x2.y; a[10]=(__bf16)x2.z; a[11]=(__bf16)x2.w;
            a[12]=(__bf16)x3.x; a[13]=(__bf16)x3.y; a[14]=(__bf16)x3.z; a[15]=(__bf16)x3.w;
        }
        // Four B fragments (32x16 each); WT rows give contiguous 16B loads.
        v16bf b[4];
        #pragma unroll
        for (int c = 0; c < 4; ++c) {
            const v8bf* bp = (const v8bf*)(Wp + (size_t)(c * 16 + nl) * EE + ko + half * 8);
            v8bf lo = bp[0], hi = bp[2];
            #pragma unroll
            for (int e = 0; e < 8; ++e) { b[c][e] = lo[e]; b[c][8 + e] = hi[e]; }
        }
        #pragma unroll
        for (int c = 0; c < 4; ++c)
            acc[c] = __builtin_amdgcn_wmma_f32_16x16x32_bf16(false, a, false, b[c],
                                                             (short)0, acc[c], false, false);

        // Pin the pipeline: all 12 VMEM reads first, then VALU converts, then WMMAs.
        SCHED_GROUP(0x020, 12, 0);   // VMEM reads (4 A + 8 B)
        SCHED_GROUP(0x002, 48, 0);   // VALU (cvt_pk_bf16 + address math)
        SCHED_GROUP(0x008,  4, 0);   // MFMA/WMMA
    }

    // C layout: VGPR r holds row M = half*8+r, col N = c*16+nl
    if (p == 2) {
        // transposed store: the 8 r-values are contiguous in s -> one 16B store
        __bf16* o = vT + (size_t)n * DD * SK;
        #pragma unroll
        for (int c = 0; c < 4; ++c) {
            v8bf pk;
            #pragma unroll
            for (int r = 0; r < 8; ++r) pk[r] = (__bf16)acc[c][r];
            *(v8bf*)(o + (size_t)(c * 16 + nl) * SK + r0 + half * 8) = pk;
        }
    } else {
        __bf16* o = (p == 0 ? qh : kh) + (size_t)n * LQ * DD;
        #pragma unroll
        for (int c = 0; c < 4; ++c)
            #pragma unroll
            for (int r = 0; r < 8; ++r)
                o[(size_t)(r0 + half * 8 + r) * DD + c * 16 + nl] = (__bf16)acc[c][r];
    }
}

// ---------------------------------------------------------------------------
// Kernel 2: causal flash attention. One wave per 16-query tile, key blocks of 32.
// scores = (qh @ khT) * 8.0  (faithful D**0.5 bug), online softmax, O += P @ V.
// ---------------------------------------------------------------------------
__global__ __launch_bounds__(256) void attn_kernel(
    const __bf16* __restrict__ qh, const __bf16* __restrict__ kh,
    const __bf16* __restrict__ vT, float* __restrict__ out)
{
    __shared__ __bf16 pbuf[8][16][40];   // per-wave P tile, 80B row stride (16B aligned)

    const int wave_g = (blockIdx.x * blockDim.x + threadIdx.x) >> 5;
    const int w    = threadIdx.x >> 5;
    const int lane = threadIdx.x & 31;
    const int nl   = lane & 15;
    const int half = lane >> 4;

    const int n  = wave_g >> 7;     // /128 query tiles
    const int qt = wave_g & 127;
    if (n >= NB) return;
    const int q0 = qt * 16;

    const __bf16* Q = qh + (size_t)n * LQ * DD;
    const __bf16* K = kh + (size_t)n * SK * DD;
    const __bf16* V = vT + (size_t)n * DD * SK;

    // Q fragments for head-dim chunks 0..31 and 32..63 (loaded once)
    v16bf aq[2];
    #pragma unroll
    for (int t = 0; t < 2; ++t) {
        const v8bf* p0 = (const v8bf*)(Q + (size_t)(q0 + nl) * DD + t * 32 + half * 16);
        const v8bf* p1 = (const v8bf*)(Q + (size_t)(q0 + nl) * DD + t * 32 + half * 16 + 8);
        v8bf lo = p0[0], hi = p1[0];
        #pragma unroll
        for (int e = 0; e < 8; ++e) { aq[t][e] = lo[e]; aq[t][8 + e] = hi[e]; }
    }

    v8f o[4] = {};                 // O[16, 64] : 4 accum tiles over d-columns
    float mrow[8], lrow[8];
    #pragma unroll
    for (int r = 0; r < 8; ++r) { mrow[r] = -__builtin_inff(); lrow[r] = 0.f; }

    const int kbmax = (q0 + 15) >> 5;
    for (int kb = 0; kb <= kbmax; ++kb) {
        const int key0 = kb * 32;
        float st[2][8];

        // ---- scores: load ALL K fragments first, then 4 WMMAs ----
        v16bf bk[2][2];
        #pragma unroll
        for (int t = 0; t < 2; ++t) {
            const int key = key0 + t * 16 + nl;
            const v8bf* kp0 = (const v8bf*)(K + (size_t)key * DD + half * 16);
            const v8bf* kp1 = (const v8bf*)(K + (size_t)key * DD + 32 + half * 16);
            v8bf x0 = kp0[0], x1 = kp0[1], y0 = kp1[0], y1 = kp1[1];
            #pragma unroll
            for (int e = 0; e < 8; ++e) {
                bk[t][0][e] = x0[e]; bk[t][0][8 + e] = x1[e];
                bk[t][1][e] = y0[e]; bk[t][1][8 + e] = y1[e];
            }
        }
        v8f s[2] = {};
        #pragma unroll
        for (int t = 0; t < 2; ++t) {
            s[t] = __builtin_amdgcn_wmma_f32_16x16x32_bf16(false, aq[0], false, bk[t][0],
                                                           (short)0, s[t], false, false);
            s[t] = __builtin_amdgcn_wmma_f32_16x16x32_bf16(false, aq[1], false, bk[t][1],
                                                           (short)0, s[t], false, false);
        }
        #pragma unroll
        for (int t = 0; t < 2; ++t) {
            const int key = key0 + t * 16 + nl;
            #pragma unroll
            for (int r = 0; r < 8; ++r) {
                float val = s[t][r] * 8.0f;                    // D**0.5 (faithful bug)
                const int qrow = q0 + half * 8 + r;
                if (key > qrow) val = -__builtin_inff();       // causal mask
                st[t][r] = val;
            }
        }

        // ---- online softmax over the 32 keys of this block ----
        #pragma unroll
        for (int r = 0; r < 8; ++r) {
            float mx = fmaxf(st[0][r], st[1][r]);
            mx = fmaxf(mx, xor_f(mx, 1)); mx = fmaxf(mx, xor_f(mx, 2));
            mx = fmaxf(mx, xor_f(mx, 4)); mx = fmaxf(mx, xor_f(mx, 8));
            const float mnew  = fmaxf(mrow[r], mx);
            const float alpha = __expf(mrow[r] - mnew);
            const float p0 = __expf(st[0][r] - mnew);
            const float p1 = __expf(st[1][r] - mnew);
            float rs = p0 + p1;
            rs += xor_f(rs, 1); rs += xor_f(rs, 2);
            rs += xor_f(rs, 4); rs += xor_f(rs, 8);
            lrow[r] = lrow[r] * alpha + rs;
            mrow[r] = mnew;
            o[0][r] *= alpha; o[1][r] *= alpha; o[2][r] *= alpha; o[3][r] *= alpha;
            pbuf[w][half * 8 + r][nl]      = (__bf16)p0;   // C-layout -> LDS
            pbuf[w][half * 8 + r][16 + nl] = (__bf16)p1;
        }

        // LDS ops from one wave are in-order; force the wait + block reordering
        __asm__ volatile("s_wait_dscnt 0" ::: "memory");

        // ---- reload P in 16-bit A layout (16x32) ----
        v16bf pa;
        {
            const v8bf* p0 = (const v8bf*)&pbuf[w][nl][half * 16];
            const v8bf* p1 = (const v8bf*)&pbuf[w][nl][half * 16 + 8];
            v8bf lo = p0[0], hi = p1[0];
            #pragma unroll
            for (int e = 0; e < 8; ++e) { pa[e] = lo[e]; pa[8 + e] = hi[e]; }
        }

        // ---- O += P @ V : load ALL V fragments first, then 4 WMMAs ----
        v16bf bv[4];
        #pragma unroll
        for (int c = 0; c < 4; ++c) {
            const v8bf* vp = (const v8bf*)(V + (size_t)(c * 16 + nl) * SK + key0 + half * 16);
            v8bf x0 = vp[0], x1 = vp[1];
            #pragma unroll
            for (int e = 0; e < 8; ++e) { bv[c][e] = x0[e]; bv[c][8 + e] = x1[e]; }
        }
        #pragma unroll
        for (int c = 0; c < 4; ++c)
            o[c] = __builtin_amdgcn_wmma_f32_16x16x32_bf16(false, pa, false, bv[c],
                                                           (short)0, o[c], false, false);
    }

    // ---- epilogue: normalize and store fp32 output ----
    #pragma unroll
    for (int r = 0; r < 8; ++r) {
        const float inv = 1.0f / lrow[r];
        const size_t row = ((size_t)n * LQ + q0 + half * 8 + r) * DD;
        out[row +  0 + nl] = o[0][r] * inv;
        out[row + 16 + nl] = o[1][r] * inv;
        out[row + 32 + nl] = o[2][r] * inv;
        out[row + 48 + nl] = o[3][r] * inv;
    }
}

extern "C" void kernel_launch(void* const* d_in, const int* in_sizes, int n_in,
                              void* d_out, int out_size, void* d_ws, size_t ws_size,
                              hipStream_t stream) {
    const float* q  = (const float*)d_in[0];
    const float* k  = (const float*)d_in[1];
    const float* v  = (const float*)d_in[2];
    const float* Wq = (const float*)d_in[3];
    const float* Wk = (const float*)d_in[4];
    const float* Wv = (const float*)d_in[5];
    // d_in[6] (attn_mask) is the known causal triu mask; applied analytically.
    float* out = (float*)d_out;

    __bf16* qh = (__bf16*)d_ws;                     // [8,2048,64] bf16 (2 MB)
    __bf16* kh = qh + (size_t)NB * LQ * DD;         // [8,2048,64] bf16 (2 MB)
    __bf16* vT = kh + (size_t)NB * SK * DD;         // [8,64,2048] bf16 (2 MB)
    __bf16* WT = vT + (size_t)NB * DD * SK;         // [3,64,1024] bf16 (384 KB)

    // Kernel order on one stream gives the needed dependencies.
    wtrans_kernel<<<(3 * EE * DD + 255) / 256, 256, 0, stream>>>(Wq, Wk, Wv, WT);
    // 3 proj * 8 batch * 128 row-tiles = 3072 waves, 8 waves/block
    proj_kernel<<<384, 256, 0, stream>>>(q, k, v, WT, qh, kh, vT);
    // 8 batch * 128 query-tiles = 1024 waves, 8 waves/block
    attn_kernel<<<128, 256, 0, stream>>>(qh, kh, vT, out);
}
